// MixedHeadsV2_21663815041814
// MI455X (gfx1250) — matrix operand order repaired and checked by reading, not compiled, not run on hardware
//
#include <hip/hip_runtime.h>
#include <hip/hip_bf16.h>
#include <math.h>

typedef _Float16 half_t;
typedef __attribute__((ext_vector_type(16))) _Float16 v16h;
typedef __attribute__((ext_vector_type(8)))  _Float16 v8h;
typedef __attribute__((ext_vector_type(8)))  float    v8f;
typedef __attribute__((ext_vector_type(4)))  float    v4f;
typedef __attribute__((ext_vector_type(4)))  int      v4i;

#define WMMA_F16(a, b, c) \
  __builtin_amdgcn_wmma_f32_16x16x32_f16(false, (a), false, (b), (short)0, (c), false, false)

// Optional CDNA5 async direct-to-LDS copy (guarded; falls back to VGPR path)
#if defined(__has_builtin)
#  if __has_builtin(__builtin_amdgcn_global_load_async_to_lds_b128)
#    define HAVE_ASYNC_LDS 1
#  endif
#  if __has_builtin(__builtin_amdgcn_s_wait_asynccnt)
#    define WAIT_ASYNC() __builtin_amdgcn_s_wait_asynccnt(0)
#  endif
#endif
#ifndef HAVE_ASYNC_LDS
#  define HAVE_ASYNC_LDS 0
#endif
#ifndef WAIT_ASYNC
#  define WAIT_ASYNC() asm volatile("s_wait_asynccnt 0x0" ::: "memory")
#endif

#if HAVE_ASYNC_LDS
typedef v4i __attribute__((address_space(1))) *g_v4i_ptr;   // prints as "v4i __device__ *"
typedef v4i __attribute__((address_space(3))) *l_v4i_ptr;   // LDS
__device__ __forceinline__ void async_copy16(const half_t* g, const half_t* l) {
  // global: full 64-bit address; LDS: addr[31:0] is the LDS offset (flat-aperture rule)
  g_v4i_ptr gp = (g_v4i_ptr)(unsigned long long)g;
  l_v4i_ptr lp = (l_v4i_ptr)(unsigned long long)(unsigned)(unsigned long long)l;
  __builtin_amdgcn_global_load_async_to_lds_b128(gp, lp, 0, 0);
}
#endif

// Problem constants
#define BB   4
#define HH   8
#define TT   2048
#define DD   128
#define CC   512
#define BT   (BB * TT)          // 8192
#define NCOL (3 * HH * DD)      // 3072

__device__ __forceinline__ v16h mk16(v8h lo, v8h hi) {
  v16h r;
#pragma unroll
  for (int i = 0; i < 8; i++) { r[i] = lo[i]; r[i + 8] = hi[i]; }
  return r;
}

// ---------------------------------------------------------------------------
// Kernel 1: fused weight-mix + QKV projection GEMM.
// Out[bt][n] = sum_c x[bt][c] * W[n][c],  W[n][c] = eff(h,d,c)*base[mat][h][d][c]
// Output written as f16 q/k/v arrays [B][H][T][D].
// ---------------------------------------------------------------------------
__global__ __launch_bounds__(256)
void mixedheads_proj_kernel(const float* __restrict__ x,
                            const float* __restrict__ wts,
                            const float* __restrict__ baseK,
                            const float* __restrict__ baseQ,
                            const float* __restrict__ baseV,
                            half_t* __restrict__ qh,
                            half_t* __restrict__ kh,
                            half_t* __restrict__ vh) {
  __shared__ __align__(16) half_t Ax[128][40];   // x tile   [m][k], stride 80B
  __shared__ __align__(16) half_t Bw[128][40];   // W tile   [n][k], stride 80B

  const int tid   = threadIdx.x;
  const int wave  = tid >> 5;
  const int lane  = tid & 31;
  const int lm    = lane & 15;
  const int lhalf = lane >> 4;
  const int mrow0 = blockIdx.x * 128;   // bt tile
  const int ncol0 = blockIdx.y * 128;   // output-column tile

  const float w0 = wts[0], w1 = wts[1], w2 = wts[2], w3 = wts[3];

  v8f zf = {};
  v8f acc[8];
#pragma unroll
  for (int i = 0; i < 8; i++) acc[i] = zf;

  // cooperative-load assignment: 256 thr = 128 rows x 2 half-rows of 16 elems
  const int lr = tid >> 1;
  const int lc = (tid & 1) * 16;

  // this thread's W row for B-tile loading
  const int   ncol = ncol0 + lr;
  const int   mat  = ncol >> 10;            // 0=Q, 1=K, 2=V
  const int   rem  = ncol & 1023;           // h*128 + d
  const int   hidx = rem >> 7;
  const int   didx = rem & 127;
  const float* basep = ((mat == 0) ? baseQ : (mat == 1) ? baseK : baseV)
                       + (size_t)rem * CC;
  const float effAll = w2 * ((hidx < 4) ? 1.f : 0.f) + w3 * ((didx < 64) ? 1.f : 0.f);
  const float effLow = w0 * (((hidx < 4) && (didx < 64)) ? 1.f : 0.f)
                     + w1 * ((didx < 32) ? 1.f : 0.f);   // only when c < 256

  for (int k0 = 0; k0 < CC; k0 += 32) {
    // stage x tile (f32 -> f16)
    {
      const float* xp = x + (size_t)(mrow0 + lr) * CC + (k0 + lc);
      if (k0 + 32 < CC) {
        __builtin_prefetch(xp + 32, 0, 1);                // next K-chunk of x
        __builtin_prefetch(basep + (k0 + 32) + lc, 0, 1); // next K-chunk of W
      }
      v4f f0 = *(const v4f*)(xp + 0);
      v4f f1 = *(const v4f*)(xp + 4);
      v4f f2 = *(const v4f*)(xp + 8);
      v4f f3 = *(const v4f*)(xp + 12);
      half_t* d = &Ax[lr][lc];
#pragma unroll
      for (int i = 0; i < 4; i++) {
        d[i + 0]  = (half_t)f0[i];
        d[i + 4]  = (half_t)f1[i];
        d[i + 8]  = (half_t)f2[i];
        d[i + 12] = (half_t)f3[i];
      }
    }
    // stage W tile (mask-mix on the fly, f32 -> f16)
    {
      const int   cb  = k0 + lc;                       // 16-chunk never crosses 256
      const float eff = effAll + ((cb < 256) ? effLow : 0.f);
      const float* bp = basep + cb;
      half_t* d = &Bw[lr][lc];
#pragma unroll
      for (int i = 0; i < 16; i++) d[i] = (half_t)(eff * bp[i]);
    }
    __syncthreads();

    // compute: each wave owns a 16-wide column strip, 8 row subtiles
    {
      const int kbB = lhalf ? 16 : 0;     // B: 16 contiguous K per lane
      const half_t* bp2 = &Bw[(wave << 4) + lm][kbB];
      v16h bfrag = mk16(*(const v8h*)bp2, *(const v8h*)(bp2 + 8));

      const int kbA = lhalf ? 16 : 0;     // A: 16 contiguous K per lane
#pragma unroll
      for (int mt = 0; mt < 8; mt++) {
        const half_t* ap = &Ax[(mt << 4) + lm][0];
        v16h afrag = mk16(*(const v8h*)(ap + kbA), *(const v8h*)(ap + kbA + 8));
        acc[mt] = WMMA_F16(afrag, bfrag, acc[mt]);
      }
    }
    __syncthreads();
  }

  // epilogue: C/D layout -> f16 q/k/v [B][H][T][D]
  const int  ncolg = ncol0 + (wave << 4) + lm;
  const int  matg  = ncolg >> 10;
  const int  remg  = ncolg & 1023;
  const int  hg    = remg >> 7;
  const int  dg    = remg & 127;
  half_t* outp = (matg == 0) ? qh : (matg == 1) ? kh : vh;
#pragma unroll
  for (int mt = 0; mt < 8; mt++) {
#pragma unroll
    for (int r = 0; r < 8; r++) {
      const int m = mrow0 + (mt << 4) + r + (lhalf << 3);
      const int b = m >> 11, t = m & (TT - 1);
      const size_t idx = (((size_t)(b * HH + hg)) * TT + t) * DD + dg;
      outp[idx] = (half_t)acc[mt][r];
    }
  }
}

// ---------------------------------------------------------------------------
// Kernel 2: causal flash attention per (b, h, 128-row query block).
// 8 waves; wave w owns query rows [w*16, w*16+16). KV streamed in blocks of 32.
// ---------------------------------------------------------------------------
__global__ __launch_bounds__(256)
void mixedheads_attn_kernel(const half_t* __restrict__ qh,
                            const half_t* __restrict__ kh,
                            const half_t* __restrict__ vh,
                            float* __restrict__ out) {
  __shared__ __align__(16) half_t Kl[32][136];      // K block  [s][d], stride 272B
  __shared__ __align__(16) half_t Vt[128][40];      // V block  [d][s], stride 80B
  __shared__ __align__(16) half_t Pl[8][16][40];    // per-wave P restripe buffer

  const int tid   = threadIdx.x;
  const int wave  = tid >> 5;
  const int lane  = tid & 31;
  const int lm    = lane & 15;
  const int lhalf = lane >> 4;
  const int kbA   = lhalf ? 16 : 0;
  const int kbB   = lhalf ? 16 : 0;

  const int qb = blockIdx.x;    // 0..15
  const int h  = blockIdx.y;    // 0..7
  const int b  = blockIdx.z;    // 0..3
  const size_t headoff = ((size_t)(b * HH + h)) * TT * DD;
  const half_t* Q = qh + headoff;
  const half_t* K = kh + headoff;
  const half_t* V = vh + headoff;

  const int t0w = qb * 128 + wave * 16;

  // Q fragments: 16 rows x 128 dims = four 16x32 A-frags, kept in registers
  v16h qf[4];
  {
    const half_t* qp = Q + (size_t)(t0w + lm) * DD;
#pragma unroll
    for (int c = 0; c < 4; c++)
      qf[c] = mk16(*(const v8h*)(qp + c * 32 + kbA),
                   *(const v8h*)(qp + c * 32 + kbA + 8));
  }

  v8f zf = {};
  v8f o[8];
#pragma unroll
  for (int i = 0; i < 8; i++) o[i] = zf;
  float mi[8], li[8];
#pragma unroll
  for (int r = 0; r < 8; r++) { mi[r] = -1e30f; li[r] = 0.f; }
  const float scale = 0.08838834764831845f;   // 1/sqrt(128)

  // cooperative KV-load assignment: 256 thr = 32 rows x 8 chunks of 16 halves
  const int ls  = tid >> 3;
  const int ld0 = (tid & 7) * 16;

  const int nblk = 4 * (qb + 1);
  for (int jb = 0; jb < nblk; jb++) {
    const int s0 = jb * 32;
    {   // K block, row-major [s][d] -- CDNA5 async copy when available
      const half_t* kp = K + (size_t)(s0 + ls) * DD + ld0;
#if HAVE_ASYNC_LDS
      async_copy16(kp,     &Kl[ls][ld0]);
      async_copy16(kp + 8, &Kl[ls][ld0 + 8]);
#else
      v8h a = *(const v8h*)kp;
      v8h c = *(const v8h*)(kp + 8);
      half_t* d = &Kl[ls][ld0];
#pragma unroll
      for (int i = 0; i < 8; i++) { d[i] = a[i]; d[8 + i] = c[i]; }
#endif
    }
    {   // V block, transposed [d][s] (needs transpose -> VGPR path)
      const half_t* vp = V + (size_t)(s0 + ls) * DD + ld0;
      v8h a = *(const v8h*)vp;
      v8h c = *(const v8h*)(vp + 8);
#pragma unroll
      for (int i = 0; i < 8; i++) { Vt[ld0 + i][ls] = a[i]; Vt[ld0 + 8 + i][ls] = c[i]; }
    }
    if (jb + 1 < nblk) {   // prefetch next KV block into cache while computing
      const half_t* kn = K + (size_t)(s0 + 32 + ls) * DD + ld0;
      const half_t* vn = V + (size_t)(s0 + 32 + ls) * DD + ld0;
      __builtin_prefetch(kn, 0, 1);
      __builtin_prefetch(vn, 0, 1);
    }
#if HAVE_ASYNC_LDS
    WAIT_ASYNC();
#endif
    __syncthreads();

    if (s0 <= t0w + 15) {   // block not entirely above this wave's diagonal
      // ---- S = Q K^T (16x32 scores), f32 accum ----
      v8f sf[2]; sf[0] = zf; sf[1] = zf;
#pragma unroll
      for (int t = 0; t < 2; t++) {
#pragma unroll
        for (int c = 0; c < 4; c++) {
          const half_t* kp = &Kl[t * 16 + lm][c * 32 + kbB];
          v16h bf = mk16(*(const v8h*)kp, *(const v8h*)(kp + 8));
          sf[t] = WMMA_F16(qf[c], bf, sf[t]);
        }
      }

      // ---- scale, causal mask, online softmax (per-row over 16-lane halves) ----
      float pv0[8], pv1[8], alpha[8];
#pragma unroll
      for (int r = 0; r < 8; r++) {
        const int tq  = t0w + r + (lhalf << 3);
        float v0 = sf[0][r] * scale; if (s0 + lm      > tq) v0 = -1e30f;
        float v1 = sf[1][r] * scale; if (s0 + 16 + lm > tq) v1 = -1e30f;
        float rm = fmaxf(v0, v1);
        rm = fmaxf(rm, __shfl_xor(rm, 1));
        rm = fmaxf(rm, __shfl_xor(rm, 2));
        rm = fmaxf(rm, __shfl_xor(rm, 4));
        rm = fmaxf(rm, __shfl_xor(rm, 8));
        const float mn = fmaxf(mi[r], rm);
        const float al = __expf(mi[r] - mn);
        const float p0 = __expf(v0 - mn);
        const float p1 = __expf(v1 - mn);
        float rs = p0 + p1;
        rs += __shfl_xor(rs, 1);
        rs += __shfl_xor(rs, 2);
        rs += __shfl_xor(rs, 4);
        rs += __shfl_xor(rs, 8);
        li[r] = li[r] * al + rs;
        mi[r] = mn;
        alpha[r] = al;
        pv0[r] = p0; pv1[r] = p1;
      }
#pragma unroll
      for (int d = 0; d < 8; d++)
#pragma unroll
        for (int r = 0; r < 8; r++) o[d][r] *= alpha[r];

      // ---- restripe P (C-layout -> A-layout) via per-wave LDS buffer ----
#pragma unroll
      for (int r = 0; r < 8; r++) {
        Pl[wave][r + (lhalf << 3)][lm]      = (half_t)pv0[r];
        Pl[wave][r + (lhalf << 3)][16 + lm] = (half_t)pv1[r];
      }
      const half_t* pp = &Pl[wave][lm][0];
      v16h pf = mk16(*(const v8h*)(pp + kbA), *(const v8h*)(pp + kbA + 8));

      // ---- O += P V ----
#pragma unroll
      for (int dsub = 0; dsub < 8; dsub++) {
        const half_t* vp2 = &Vt[dsub * 16 + lm][kbB];
        v16h bf = mk16(*(const v8h*)vp2, *(const v8h*)(vp2 + 8));
        o[dsub] = WMMA_F16(pf, bf, o[dsub]);
      }
    }
    __syncthreads();
  }

  // ---- normalize and write out[b][t][h*128 + d] (f32) ----
  float inv[8];
#pragma unroll
  for (int r = 0; r < 8; r++) inv[r] = 1.0f / li[r];
#pragma unroll
  for (int dsub = 0; dsub < 8; dsub++) {
#pragma unroll
    for (int r = 0; r < 8; r++) {
      const int tq  = t0w + r + (lhalf << 3);
      const int col = h * DD + dsub * 16 + lm;
      out[((size_t)b * TT + tq) * (HH * DD) + col] = o[dsub][r] * inv[r];
    }
  }
}

extern "C" void kernel_launch(void* const* d_in, const int* in_sizes, int n_in,
                              void* d_out, int out_size, void* d_ws, size_t ws_size,
                              hipStream_t stream) {
  (void)in_sizes; (void)n_in; (void)out_size; (void)ws_size;
  const float* x     = (const float*)d_in[0];
  const float* wts   = (const float*)d_in[1];
  const float* baseK = (const float*)d_in[2];
  const float* baseQ = (const float*)d_in[3];
  const float* baseV = (const float*)d_in[4];
  float* out = (float*)d_out;

  const size_t elems = (size_t)BB * HH * TT * DD;   // 8,388,608 halves each
  half_t* qh = (half_t*)d_ws;
  half_t* kh = qh + elems;
  half_t* vh = kh + elems;

  // 1) fused weight-mix + QKV projection: 8192 x 3072 x 512 GEMM
  dim3 pgrid(BT / 128, NCOL / 128);     // 64 x 24
  mixedheads_proj_kernel<<<pgrid, 256, 0, stream>>>(x, wts, baseK, baseQ, baseV,
                                                    qh, kh, vh);

  // 2) causal flash attention, one WG per (b, h, 128 query rows)
  dim3 agrid(TT / 128, HH, BB);         // 16 x 8 x 4
  mixedheads_attn_kernel<<<agrid, 256, 0, stream>>>(qh, kh, vh, out);
}
